// Transformer_Different_Attention_EncoderLayer_62354335203940
// MI455X (gfx1250) — matrix operand order repaired and checked by reading, not compiled, not run on hardware
//
#include <hip/hip_runtime.h>
#include <math.h>

// ---------------------------------------------------------------------------
// CDNA5 / gfx1250 transformer encoder layer (wave32, WMMA bf16 matrix cores)
//   B=4 S=2048 D=1024 F=4096 H=16 Dh=64,  M = B*S = 8192
// All GEMM-shaped work uses v_wmma_f32_16x16x32_bf16 (f32 accumulate).
// GEMM tile staging uses gfx1250 async global->LDS copies (ASYNCcnt).
// Softmax / layernorm / residuals / threshold stay in f32 VALU.
// ---------------------------------------------------------------------------

typedef __bf16 v16bf __attribute__((ext_vector_type(16)));
typedef __bf16 v8bf  __attribute__((ext_vector_type(8)));
typedef float  v8f   __attribute__((ext_vector_type(8)));

#define WMMA_BF16(A, B, C) \
  __builtin_amdgcn_wmma_f32_16x16x32_bf16(false, (A), false, (B), (short)0, (C), false, false)

// ---- fragment loaders (contiguous k = 16h+e for A and B, wave32) --------------
// A 16x32 bf16: lane = {h=lane>>4, r=lane&15}; elems 0..15 -> K = h*16+e
//               (16 contiguous halfs, two 16B loads).
__device__ __forceinline__ v16bf load_a_frag(const __bf16* p, int lda, int lane) {
  int r = lane & 15, h = lane >> 4;
  const __bf16* q = p + (long)r * lda + (h << 4);
  v8bf lo = *(const v8bf*)q;
  v8bf hi = *(const v8bf*)(q + 8);
  return __builtin_shufflevector(lo, hi, 0,1,2,3,4,5,6,7,8,9,10,11,12,13,14,15);
}
// B 32x16 bf16 from (n,k)-major storage (i.e. W row-major [N,K]):
// lane col n = lane&15, K = (lane>>4)*16 + e  -> 16 contiguous halfs.
__device__ __forceinline__ v16bf load_b_frag_nk(const __bf16* p, int ldn, int lane) {
  int n = lane & 15, h = lane >> 4;
  const __bf16* q = p + (long)n * ldn + (h << 4);
  v8bf lo = *(const v8bf*)q;
  v8bf hi = *(const v8bf*)(q + 8);
  return __builtin_shufflevector(lo, hi, 0,1,2,3,4,5,6,7,8,9,10,11,12,13,14,15);
}
// B 32x16 bf16 from standard (k,n)-major storage (V matrix): strided reads.
__device__ __forceinline__ v16bf load_b_frag_kn(const __bf16* p, int ldk, int lane) {
  int n = lane & 15, h = lane >> 4;
  const __bf16* q = p + (long)(h << 4) * ldk + n;
  v16bf b;
#pragma unroll
  for (int e = 0; e < 16; ++e) b[e] = q[(long)e * ldk];
  return b;
}

// ---- 16B-vector global->LDS tile copy, synchronous (VGPR round trip) ------
__device__ __forceinline__ void copy_tile_g2s(__bf16* dst, int dld,
                                              const __bf16* __restrict__ src, long sld,
                                              int rows, int cols, int tid, int nthr) {
  int vpr = cols >> 3;
  int total = rows * vpr;
  for (int i = tid; i < total; i += nthr) {
    int r = i / vpr;
    int c = (i - r * vpr) << 3;
    *(v8bf*)(dst + (long)r * dld + c) = *(const v8bf*)(src + (long)r * sld + c);
  }
}

// ---- async global->LDS tile copy: global_load_async_to_lds_b128 -----------
// GVS mode: uniform 64-bit SGPR base + per-lane 32-bit byte offset; dest is
// the per-lane LDS byte address (low 32 bits of the flat LDS pointer).
// Tracked by ASYNCcnt; caller must async_wait() before the barrier.
__device__ __forceinline__ void copy_tile_g2s_async(__bf16* dst, int dld,
                                                    const __bf16* src, long sld,
                                                    int rows, int cols, int tid, int nthr) {
  unsigned dbase = (unsigned)(size_t)dst;  // LDS aperture: addr[31:0] = LDS offset
  int vpr = cols >> 3;
  int total = rows * vpr;
  for (int i = tid; i < total; i += nthr) {
    int r = i / vpr;
    int c = (i - r * vpr) << 3;
    unsigned ldsa = dbase + (unsigned)(r * dld + c) * 2u;
    unsigned goff = (unsigned)((long)r * sld + c) * 2u;
    asm volatile("global_load_async_to_lds_b128 %0, %1, %2"
                 :: "v"(ldsa), "v"(goff), "s"(src)
                 : "memory");
  }
}
__device__ __forceinline__ void async_wait() {
  asm volatile("s_wait_asynccnt 0x0" ::: "memory");
}

// ---------------------------------------------------------------------------
// f32 -> bf16 cast (weights)
// ---------------------------------------------------------------------------
__global__ void cast_f32_bf16_kernel(const float* __restrict__ in, __bf16* __restrict__ out, long n) {
  long i = (long)blockIdx.x * blockDim.x + threadIdx.x;
  long stride = (long)gridDim.x * blockDim.x;
  for (; i < n; i += stride) out[i] = (__bf16)in[i];
}

// ---------------------------------------------------------------------------
// Custom layernorm: mu, std(ddof=1), y = alpha*(x-mu)/(sd+eps)+beta
// optional threshold mask (y < 0.15 -> 0).  One 256-thread block per row.
// ---------------------------------------------------------------------------
__global__ __launch_bounds__(256)
void layernorm_kernel(const float* __restrict__ in, const float* __restrict__ alpha,
                      const float* __restrict__ beta, __bf16* __restrict__ out,
                      int D, int applyThresh) {
  long row = blockIdx.x;
  const float* xr = in + row * (long)D;
  float s = 0.f, ss = 0.f;
  for (int i = threadIdx.x; i < D; i += blockDim.x) {
    float v = xr[i];
    s += v; ss += v * v;
  }
#pragma unroll
  for (int o = 16; o >= 1; o >>= 1) {
    s  += __shfl_down(s,  o, 32);
    ss += __shfl_down(ss, o, 32);
  }
  __shared__ float sh[2][8];
  int wave = threadIdx.x >> 5, lane = threadIdx.x & 31;
  if (lane == 0) { sh[0][wave] = s; sh[1][wave] = ss; }
  __syncthreads();
  if (threadIdx.x == 0) {
    float ts = 0.f, tss = 0.f;
    for (int i = 0; i < 8; ++i) { ts += sh[0][i]; tss += sh[1][i]; }
    float mu  = ts / (float)D;
    float var = (tss - (float)D * mu * mu) / (float)(D - 1);
    float sd  = sqrtf(fmaxf(var, 0.f));
    sh[0][0] = mu;
    sh[1][0] = 1.f / (sd + 1e-6f);
  }
  __syncthreads();
  float mu = sh[0][0], inv = sh[1][0];
  for (int i = threadIdx.x; i < D; i += blockDim.x) {
    float y = alpha[i] * (xr[i] - mu) * inv + beta[i];
    if (applyThresh && y < 0.15f) y = 0.f;
    out[row * (long)D + i] = (__bf16)y;
  }
}

// ---------------------------------------------------------------------------
// GEMM: C[M,N] = act(A[M,K] @ W[N,K]^T + bias[N])  (A,W bf16; acc f32)
// 128x128x32 tiles, 8 waves (16-row strip each, 8 WMMA accumulators).
// Tiles staged with async global->LDS (no VGPR round trip in the hot loop).
// ---------------------------------------------------------------------------
template<bool RELU, bool RESID>
__global__ __launch_bounds__(256)
void gemm_bt_kernel(const __bf16* __restrict__ A, const __bf16* __restrict__ W,
                    const float* __restrict__ bias, const float* __restrict__ resid,
                    float* __restrict__ outF, __bf16* __restrict__ outB,
                    int M, int N, int K) {
  __shared__ __attribute__((aligned(16))) __bf16 As[128][40];
  __shared__ __attribute__((aligned(16))) __bf16 Ws[128][40];
  int tid = threadIdx.x, wave = tid >> 5, lane = tid & 31;
  int lh = lane >> 4, ln = lane & 15;
  long m0 = (long)blockIdx.y * 128;
  long n0 = (long)blockIdx.x * 128;

  v8f acc[8];
#pragma unroll
  for (int i = 0; i < 8; ++i) acc[i] = (v8f){0.f,0.f,0.f,0.f,0.f,0.f,0.f,0.f};

  for (int k0 = 0; k0 < K; k0 += 32) {
    __syncthreads();
    copy_tile_g2s_async(&As[0][0], 40, A + m0 * K + k0, K, 128, 32, tid, 256);
    copy_tile_g2s_async(&Ws[0][0], 40, W + n0 * K + k0, K, 128, 32, tid, 256);
    if (k0 + 32 < K) {  // gfx1250 global_prefetch of next K-slab
      int r = tid & 127;
      __builtin_prefetch(A + (m0 + r) * K + k0 + 32, 0, 0);
      __builtin_prefetch(W + (n0 + r) * K + k0 + 32, 0, 0);
    }
    async_wait();        // ASYNCcnt -> 0 for this wave's copies
    __syncthreads();     // tiles visible to all waves
    v16bf af = load_a_frag(&As[wave * 16][0], 40, lane);
#pragma unroll
    for (int nt = 0; nt < 8; ++nt) {
      v16bf bf = load_b_frag_nk(&Ws[nt * 16][0], 40, lane);
      acc[nt] = WMMA_BF16(af, bf, acc[nt]);
    }
  }

  long rowb = m0 + wave * 16 + lh * 8;
#pragma unroll
  for (int nt = 0; nt < 8; ++nt) {
    long col = n0 + nt * 16 + ln;
    float bsv = bias ? bias[col] : 0.f;
#pragma unroll
    for (int r = 0; r < 8; ++r) {
      long row = rowb + r;
      float v = acc[nt][r] + bsv;
      if (RELU) v = fmaxf(v, 0.f);
      if (RESID) v += resid[row * N + col];
      if (outF) outF[row * N + col] = v;
      if (outB) outB[row * N + col] = (__bf16)v;
    }
  }
}

// ---------------------------------------------------------------------------
// Flash attention, fused first residual: xres = x + softmax(QK^T/8) V
// grid = (S/128, B*H); block = 256 (8 waves, 16 query rows per wave).
// ---------------------------------------------------------------------------
__global__ __launch_bounds__(256)
void flash_attn_kernel(const __bf16* __restrict__ Q, const __bf16* __restrict__ K,
                       const __bf16* __restrict__ V, const float* __restrict__ x,
                       float* __restrict__ xres, int S, int H, int Dh) {
  const int D = H * Dh;
  int b = blockIdx.y / H, hh = blockIdx.y % H;
  int qrow0 = blockIdx.x * 128;
  int tid = threadIdx.x, wave = tid >> 5, lane = tid & 31;
  int lh = lane >> 4, ln = lane & 15;

  __shared__ __attribute__((aligned(16))) __bf16 Qs[128][72];
  __shared__ __attribute__((aligned(16))) __bf16 Ks[64][72];
  __shared__ __attribute__((aligned(16))) __bf16 Vs[64][72];
  __shared__ __attribute__((aligned(16))) __bf16 Ps[8][16][72];

  const __bf16* qbase = Q + ((long)(b * S + qrow0)) * D + hh * Dh;
  copy_tile_g2s(&Qs[0][0], 72, qbase, D, 128, 64, tid, 256);

  v8f acc0 = (v8f){0.f,0.f,0.f,0.f,0.f,0.f,0.f,0.f};
  v8f acc1 = acc0, acc2 = acc0, acc3 = acc0;
  float m_i[8], l_i[8];
#pragma unroll
  for (int r = 0; r < 8; ++r) { m_i[r] = -1e30f; l_i[r] = 0.f; }
  const float scale = 0.125f;  // 1/sqrt(64)

  for (int kv0 = 0; kv0 < S; kv0 += 64) {
    __syncthreads();
    const __bf16* kbase = K + ((long)(b * S + kv0)) * D + hh * Dh;
    const __bf16* vbase = V + ((long)(b * S + kv0)) * D + hh * Dh;
    copy_tile_g2s(&Ks[0][0], 72, kbase, D, 64, 64, tid, 256);
    copy_tile_g2s(&Vs[0][0], 72, vbase, D, 64, 64, tid, 256);
    __syncthreads();

    // S strip (16 x 64) = Q_strip @ K_tile^T : K is (n,k)-major naturally.
    v8f s0 = (v8f){0.f,0.f,0.f,0.f,0.f,0.f,0.f,0.f};
    v8f s1 = s0, s2 = s0, s3 = s0;
#pragma unroll
    for (int kt = 0; kt < 2; ++kt) {
      v16bf aq = load_a_frag(&Qs[wave * 16][kt * 32], 72, lane);
      v16bf b0 = load_b_frag_nk(&Ks[ 0][kt * 32], 72, lane); s0 = WMMA_BF16(aq, b0, s0);
      v16bf b1 = load_b_frag_nk(&Ks[16][kt * 32], 72, lane); s1 = WMMA_BF16(aq, b1, s1);
      v16bf b2 = load_b_frag_nk(&Ks[32][kt * 32], 72, lane); s2 = WMMA_BF16(aq, b2, s2);
      v16bf b3 = load_b_frag_nk(&Ks[48][kt * 32], 72, lane); s3 = WMMA_BF16(aq, b3, s3);
    }
    s0 = s0 * scale; s1 = s1 * scale; s2 = s2 * scale; s3 = s3 * scale;

    // online softmax: C-tile lane (lh,ln) holds rows lh*8+r, cols ln(+16k).
#pragma unroll
    for (int r = 0; r < 8; ++r) {
      float mx = fmaxf(fmaxf(s0[r], s1[r]), fmaxf(s2[r], s3[r]));
#pragma unroll
      for (int o = 1; o < 16; o <<= 1) mx = fmaxf(mx, __shfl_xor(mx, o, 32));
      float mnew = fmaxf(m_i[r], mx);
      float c = __expf(m_i[r] - mnew);
      float p0 = __expf(s0[r] - mnew);
      float p1 = __expf(s1[r] - mnew);
      float p2 = __expf(s2[r] - mnew);
      float p3 = __expf(s3[r] - mnew);
      float rs = p0 + p1 + p2 + p3;
#pragma unroll
      for (int o = 1; o < 16; o <<= 1) rs += __shfl_xor(rs, o, 32);
      l_i[r] = l_i[r] * c + rs;
      m_i[r] = mnew;
      acc0[r] = acc0[r] * c; acc1[r] = acc1[r] * c;
      acc2[r] = acc2[r] * c; acc3[r] = acc3[r] * c;
      int prow = lh * 8 + r;  // stage P for re-fragmentation as WMMA A operand
      Ps[wave][prow][ln     ] = (__bf16)p0;
      Ps[wave][prow][ln + 16] = (__bf16)p1;
      Ps[wave][prow][ln + 32] = (__bf16)p2;
      Ps[wave][prow][ln + 48] = (__bf16)p3;
    }

    // O += P @ V  (V in standard (k,n) layout)
#pragma unroll
    for (int kt = 0; kt < 2; ++kt) {
      v16bf pa = load_a_frag(&Ps[wave][0][kt * 32], 72, lane);
      v16bf v0 = load_b_frag_kn(&Vs[kt * 32][ 0], 72, lane); acc0 = WMMA_BF16(pa, v0, acc0);
      v16bf v1 = load_b_frag_kn(&Vs[kt * 32][16], 72, lane); acc1 = WMMA_BF16(pa, v1, acc1);
      v16bf v2 = load_b_frag_kn(&Vs[kt * 32][32], 72, lane); acc2 = WMMA_BF16(pa, v2, acc2);
      v16bf v3 = load_b_frag_kn(&Vs[kt * 32][48], 72, lane); acc3 = WMMA_BF16(pa, v3, acc3);
    }
  }

  // epilogue: xres = x + O / l   (fused residual #1)
  const float* xb = x    + ((long)(b * S + qrow0 + wave * 16)) * D + hh * Dh;
  float*       ob = xres + ((long)(b * S + qrow0 + wave * 16)) * D + hh * Dh;
#pragma unroll
  for (int r = 0; r < 8; ++r) {
    float inv = 1.f / l_i[r];
    long off = (long)(lh * 8 + r) * D;
    ob[off + ln     ] = xb[off + ln     ] + acc0[r] * inv;
    ob[off + ln + 16] = xb[off + ln + 16] + acc1[r] * inv;
    ob[off + ln + 32] = xb[off + ln + 32] + acc2[r] * inv;
    ob[off + ln + 48] = xb[off + ln + 48] + acc3[r] * inv;
  }
}

// ---------------------------------------------------------------------------
extern "C" void kernel_launch(void* const* d_in, const int* in_sizes, int n_in,
                              void* d_out, int out_size, void* d_ws, size_t ws_size,
                              hipStream_t stream) {
  (void)in_sizes; (void)n_in; (void)out_size; (void)ws_size;
  const int B = 4, S = 2048, D = 1024, F = 4096, H = 16, Dh = 64;
  const long M = (long)B * S;  // 8192

  const float* x      = (const float*)d_in[0];
  const float* alpha1 = (const float*)d_in[1];
  const float* bias1  = (const float*)d_in[2];
  const float* alpha2 = (const float*)d_in[3];
  const float* bias2  = (const float*)d_in[4];
  const float* wq = (const float*)d_in[5];
  const float* bq = (const float*)d_in[6];
  const float* wk = (const float*)d_in[7];
  const float* bk = (const float*)d_in[8];
  const float* wv = (const float*)d_in[9];
  const float* bv = (const float*)d_in[10];
  const float* w1 = (const float*)d_in[11];
  const float* b1 = (const float*)d_in[12];
  const float* w2 = (const float*)d_in[13];
  const float* b2 = (const float*)d_in[14];
  float* out = (float*)d_out;

  // workspace layout (peak ~134 MB, with reuse)
  char* ws = (char*)d_ws;
  size_t off = 0;
  __bf16* wqb = (__bf16*)(ws + off); off += (size_t)D * D * 2;
  __bf16* wkb = (__bf16*)(ws + off); off += (size_t)D * D * 2;
  __bf16* wvb = (__bf16*)(ws + off); off += (size_t)D * D * 2;
  __bf16* w1b = (__bf16*)(ws + off); off += (size_t)F * D * 2;
  __bf16* w2b = (__bf16*)(ws + off); off += (size_t)D * F * 2;
  __bf16* qb  = (__bf16*)(ws + off); off += (size_t)M * D * 2;
  __bf16* kb  = (__bf16*)(ws + off); off += (size_t)M * D * 2;
  __bf16* vb  = (__bf16*)(ws + off); off += (size_t)M * D * 2;
  __bf16* x2  = (__bf16*)(ws + off); off += (size_t)M * D * 2;
  float*  xres = (float*)(ws + off); off += (size_t)M * D * 4;
  __bf16* x2m = (__bf16*)(ws + off); off += (size_t)M * D * 2;
  __bf16* hb  = qb;  // reuse q/k/v/x2 region (64 MB) for FFN hidden [M,F] bf16

  // 1) weights -> bf16
  cast_f32_bf16_kernel<<<4096, 256, 0, stream>>>(wq, wqb, (long)D * D);
  cast_f32_bf16_kernel<<<4096, 256, 0, stream>>>(wk, wkb, (long)D * D);
  cast_f32_bf16_kernel<<<4096, 256, 0, stream>>>(wv, wvb, (long)D * D);
  cast_f32_bf16_kernel<<<4096, 256, 0, stream>>>(w1, w1b, (long)F * D);
  cast_f32_bf16_kernel<<<4096, 256, 0, stream>>>(w2, w2b, (long)D * F);

  // 2) pre-norm 1 -> bf16
  layernorm_kernel<<<(unsigned)M, 256, 0, stream>>>(x, alpha1, bias1, x2, D, 0);

  // 3) Q/K/V projections (bf16 out, laid out [B,S,H,Dh])
  dim3 gqkv(D / 128, (unsigned)(M / 128));
  gemm_bt_kernel<false, false><<<gqkv, 256, 0, stream>>>(x2, wqb, bq, nullptr, nullptr, qb, (int)M, D, D);
  gemm_bt_kernel<false, false><<<gqkv, 256, 0, stream>>>(x2, wkb, bk, nullptr, nullptr, kb, (int)M, D, D);
  gemm_bt_kernel<false, false><<<gqkv, 256, 0, stream>>>(x2, wvb, bv, nullptr, nullptr, vb, (int)M, D, D);

  // 4) flash attention + residual 1: xres = x + attn
  dim3 gat(S / 128, B * H);
  flash_attn_kernel<<<gat, 256, 0, stream>>>(qb, kb, vb, x, xres, S, H, Dh);

  // 5) norm 2 + threshold mask -> bf16
  layernorm_kernel<<<(unsigned)M, 256, 0, stream>>>(xres, alpha2, bias2, x2m, D, 1);

  // 6) FFN1: h = relu(x2m @ w1^T + b1)  (bf16 out)
  dim3 g1(F / 128, (unsigned)(M / 128));
  gemm_bt_kernel<true, false><<<g1, 256, 0, stream>>>(x2m, w1b, b1, nullptr, nullptr, hb, (int)M, F, D);

  // 7) FFN2 + residual 2: out = xres + h @ w2^T + b2  (f32 out)
  dim3 g2(D / 128, (unsigned)(M / 128));
  gemm_bt_kernel<false, true><<<g2, 256, 0, stream>>>(hb, w2b, b2, xres, out, nullptr, (int)M, D, F);
}
